// ImprovedEGNNLayer_6975026888916
// MI455X (gfx1250) — matrix operand order repaired and checked by reading, not compiled, not run on hardware
//
#include <hip/hip_runtime.h>

typedef __bf16 v16bf __attribute__((ext_vector_type(16)));
typedef __bf16 v8bf  __attribute__((ext_vector_type(8)));
typedef float  v8f   __attribute__((ext_vector_type(8)));

__device__ __forceinline__ float leakyf(float x) { return x > 0.0f ? x : 0.1f * x; }

__device__ __forceinline__ v8f wmma_bf16(v16bf a, v16bf b, v8f c) {
    return __builtin_amdgcn_wmma_f32_16x16x32_bf16(
        false, a, false, b, (short)0, c, false, false);
}

// A-fragment: lanes 0-15 hold K = k0+{0..7,16..23}, lanes 16-31 K = k0+{8..15,24..31}.
__device__ __forceinline__ v16bf ldA(const __bf16* rowp, int k0, int half) {
    v8bf lo = *(const v8bf*)(rowp + k0 + (half << 3));
    v8bf hi = *(const v8bf*)(rowp + k0 + 16 + (half << 3));
    return __builtin_shufflevector(lo, hi, 0,1,2,3,4,5,6,7,8,9,10,11,12,13,14,15);
}
__device__ __forceinline__ v16bf ldA_cvt(const float* rowp, int k0, int half) {
    const float* p0 = rowp + k0 + (half << 3);
    const float* p1 = rowp + k0 + 16 + (half << 3);
    v16bf a;
#pragma unroll
    for (int e = 0; e < 8; ++e) { a[e] = (__bf16)p0[e]; a[e + 8] = (__bf16)p1[e]; }
    return a;
}
// B-fragment from an LDS stage buffer: s[chunk*512 + nl*32 + t], two 16B reads (t = 8h.., 16+8h..).
__device__ __forceinline__ v16bf ldB_lds(const __bf16* s, int chunk, int nl, int half) {
    const __bf16* p = s + (chunk << 9) + (nl << 5) + (half << 3);
    v8bf lo = *(const v8bf*)(p);
    v8bf hi = *(const v8bf*)(p + 16);
    return __builtin_shufflevector(lo, hi, 0,1,2,3,4,5,6,7,8,9,10,11,12,13,14,15);
}

// Issue async copies of one 16-column weight slice (nt) into LDS (no wait here).
// Global swizzled layout: W[((chunk*256 + n)*32) + t]; per (chunk,nt) region is
// 512 contiguous bf16 (1KB). chunks*64 16B packets spread over 256 threads:
// each *wave* issues (chunks/4) async instructions -> ASYNCcnt += chunks/4.
__device__ __forceinline__ void stage_issue(__bf16* dst, const __bf16* w, int nt,
                                            int chunks, int tid) {
    int packets = chunks << 6;
    for (int p = tid; p < packets; p += 256) {
        int c = p >> 6;
        int q = p & 63;
        const __bf16* g = w + ((((c << 8) + (nt << 4))) << 5) + (q << 3);
        unsigned lofs = (unsigned)(size_t)(dst + (c << 9) + (q << 3));
        asm volatile("global_load_async_to_lds_b128 %0, %1, off"
                     :: "v"(lofs), "v"(g) : "memory");
    }
}
// Double-buffered pipeline step: issue slice nt+1, then wait for slice nt.
// ASYNC loads complete in order, so waiting ASYNCcnt <= (packets just issued)
// guarantees the previous slice has fully landed.
template <int CH>
__device__ __forceinline__ void stage_next_and_wait(__bf16 (*bufs)[16 * 512],
                                                    const __bf16* w, int nt,
                                                    int ntEnd, int tid) {
    if (nt + 1 < ntEnd) {
        stage_issue(bufs[(nt + 1) & 1], w, nt + 1, CH, tid);
        if (CH == 16) asm volatile("s_wait_asynccnt 0x4" ::: "memory");
        else          asm volatile("s_wait_asynccnt 0x2" ::: "memory");
    } else {
        asm volatile("s_wait_asynccnt 0x0" ::: "memory");
    }
}

// ---------------------------------------------------------------- prep kernels
__global__ void k_zero_f32(float* p, int n) {
    int i = blockIdx.x * blockDim.x + threadIdx.x;
    if (i < n) p[i] = 0.0f;
}
__global__ void k_copy_f32(float* dst, const float* src, int n) {
    int i = blockIdx.x * blockDim.x + threadIdx.x;
    if (i < n) dst[i] = src[i];
}
__global__ void k_cvt_bf16(__bf16* dst, const float* src, int n) {
    int i = blockIdx.x * blockDim.x + threadIdx.x;
    if (i < n) dst[i] = (__bf16)src[i];
}
__global__ void k_swz(__bf16* dst, const float* src, int chunks) {
    int idx = blockIdx.x * blockDim.x + threadIdx.x;
    int total = chunks << 13;
    if (idx >= total) return;
    int c = idx >> 13;
    int rem = idx & 8191;
    int n = rem >> 5;
    int t = rem & 31;
    dst[idx] = (__bf16)src[(size_t)((c << 5) + t) * 256 + n];
}

// ---------------------------------------------------------------- edge kernel
// 256 threads = 8 waves; each wave owns 16 edges (128 edges / workgroup).
// B slices double-buffered in LDS via async-to-LDS; one shared activation buffer
// (GEMM2 overwrites GEMM1's slice after register preload). ~99KB LDS -> 3 WG/WGP.
__global__ __launch_bounds__(256) void egnn_edge(
    const float* __restrict__ coords,
    const int* __restrict__ rowI, const int* __restrict__ colI,
    const __bf16* __restrict__ hbf,
    const __bf16* __restrict__ w1s, const float* __restrict__ ew1_raw,
    const float* __restrict__ b1,
    const __bf16* __restrict__ w2s, const float* __restrict__ b2,
    const float* __restrict__ g, const float* __restrict__ beta,
    const __bf16* __restrict__ cw1s, const float* __restrict__ cb1,
    const float* __restrict__ cw2, const float* __restrict__ cscale,
    float* __restrict__ agg, float* __restrict__ coordsOut, int E)
{
    __shared__ __bf16 sStage[2][16 * 512];     // 32KB: double-buffered B slices
    __shared__ __bf16 sAct[8][16 * 256];       // 64KB: per-wave activation tile
    __shared__ int    sR[128], sC[128];
    __shared__ float  sRel[128][3], sDist[128];

    const int tid  = threadIdx.x;
    const int wid  = tid >> 5;
    const int lane = tid & 31;
    const int lm   = lane & 15;
    const int half = lane >> 4;
    const int tileBase = blockIdx.x * 128;
    const int waveBase = tileBase + wid * 16;

    if (tid < 128) {
        int e = tileBase + tid;
        int r = 0, c = 0;
        float rx = 0.f, ry = 0.f, rz = 0.f, d = 1.0f;
        if (e < E) {
            r = rowI[e]; c = colI[e];
            rx = coords[r * 3 + 0] - coords[c * 3 + 0];
            ry = coords[r * 3 + 1] - coords[c * 3 + 1];
            rz = coords[r * 3 + 2] - coords[c * 3 + 2];
            d = sqrtf(rx * rx + ry * ry + rz * rz) + 1e-8f;
        }
        sR[tid] = r; sC[tid] = c;
        sRel[tid][0] = rx; sRel[tid][1] = ry; sRel[tid][2] = rz;
        sDist[tid] = d;
    }
    __syncthreads();

    // All 16 A-chunk fragments register-resident across the whole GEMM1 loop.
    const __bf16* arow0 = hbf + (size_t)sR[wid * 16 + lm] * 256;
    const __bf16* arow1 = hbf + (size_t)sC[wid * 16 + lm] * 256;
    v16bf aF[16];
#pragma unroll
    for (int c = 0; c < 8; ++c) {
        aF[c]     = ldA(arow0, c * 32, half);
        aF[8 + c] = ldA(arow1, c * 32, half);
    }

    // ---- GEMM1: [h[row] | h[col] | dist] @ e_w1 + b1, leaky -> sAct (bf16)
    stage_issue(sStage[0], w1s, 0, 16, tid);
    for (int nt = 0; nt < 16; ++nt) {
        stage_next_and_wait<16>(sStage, w1s, nt, 16, tid);
        __syncthreads();                       // slice nt visible to all waves
        const __bf16* buf = sStage[nt & 1];
        v8f acc = {0.f,0.f,0.f,0.f,0.f,0.f,0.f,0.f};
#pragma unroll
        for (int c = 0; c < 16; ++c)
            acc = wmma_bf16(aF[c], ldB_lds(buf, c, lm, half), acc);
        int n = nt * 16 + lm;
        float bb = b1[n];
        float wl = ew1_raw[512 * 256 + n];     // rank-1 dist column
#pragma unroll
        for (int j = 0; j < 8; ++j) {
            int m = half * 8 + j;
            sAct[wid][m * 256 + n] =
                (__bf16)leakyf(acc[j] + bb + sDist[wid * 16 + m] * wl);
        }
        __syncthreads();                       // slice nt consumed by all waves
    }

    // Preload GEMM2 A-fragments (wave-local slice -> registers); sAct then reusable.
    v16bf aF2[8];
    {
        const __bf16* arow = &sAct[wid][lm * 256];
#pragma unroll
        for (int c = 0; c < 8; ++c) aF2[c] = ldA(arow, c * 32, half);
    }

    // ---- GEMM2: x @ e_w2 + b2, leaky -> sAct (overwrite, safe after preload)
    stage_issue(sStage[0], w2s, 0, 8, tid);
    for (int nt = 0; nt < 16; ++nt) {
        stage_next_and_wait<8>(sStage, w2s, nt, 16, tid);
        __syncthreads();
        const __bf16* buf = sStage[nt & 1];
        v8f acc = {0.f,0.f,0.f,0.f,0.f,0.f,0.f,0.f};
#pragma unroll
        for (int c = 0; c < 8; ++c)
            acc = wmma_bf16(aF2[c], ldB_lds(buf, c, lm, half), acc);
        int n = nt * 16 + lm;
        float bb = b2[n];
#pragma unroll
        for (int j = 0; j < 8; ++j) {
            int m = half * 8 + j;
            sAct[wid][m * 256 + n] = (__bf16)leakyf(acc[j] + bb);
        }
        __syncthreads();
    }

    // ---- LayerNorm rows (wave-local) -> edge_attr in sAct
    if (lane < 16) {
        __bf16* rowp = &sAct[wid][lane * 256];
        float mu = 0.f;
        for (int c2 = 0; c2 < 256; ++c2) mu += (float)rowp[c2];
        mu *= (1.0f / 256.0f);
        float var = 0.f;
        for (int c2 = 0; c2 < 256; ++c2) {
            float d = (float)rowp[c2] - mu; var += d * d;
        }
        var *= (1.0f / 256.0f);
        float rstd = rsqrtf(var + 1e-5f);
        for (int c2 = 0; c2 < 256; ++c2)
            rowp[c2] = (__bf16)(((float)rowp[c2] - mu) * rstd * g[c2] + beta[c2]);
    }

    // ---- scatter edge_attr into agg_attr[row] (wave-local source)
    {
        const __bf16* myAct = &sAct[wid][0];
        for (int idx = lane; idx < 16 * 256; idx += 32) {
            int m = idx >> 8;
            if (waveBase + m < E)
                atomicAdd(&agg[(size_t)sR[wid * 16 + m] * 256 + (idx & 255)],
                          (float)myAct[idx]);
        }
    }

    // Preload GEMM3 A-fragments (normalized edge_attr).
    v16bf aF3[8];
    {
        const __bf16* arow = &sAct[wid][lm * 256];
#pragma unroll
        for (int c = 0; c < 8; ++c) aF3[c] = ldA(arow, c * 32, half);
    }

    // ---- GEMM3 + folded c_w2 dot: cm[m] = sum_n leaky(.)*c_w2[n] (registers only)
    float cmPart[8] = {0.f,0.f,0.f,0.f,0.f,0.f,0.f,0.f};
    stage_issue(sStage[0], cw1s, 0, 8, tid);
    for (int nt = 0; nt < 16; ++nt) {
        stage_next_and_wait<8>(sStage, cw1s, nt, 16, tid);
        __syncthreads();
        const __bf16* buf = sStage[nt & 1];
        v8f acc = {0.f,0.f,0.f,0.f,0.f,0.f,0.f,0.f};
#pragma unroll
        for (int c = 0; c < 8; ++c)
            acc = wmma_bf16(aF3[c], ldB_lds(buf, c, lm, half), acc);
        int n = nt * 16 + lm;
        float bb = cb1[n];
        float wv = cw2[n];
#pragma unroll
        for (int j = 0; j < 8; ++j)
            cmPart[j] += leakyf(acc[j] + bb) * wv;
        __syncthreads();
    }
    // Reduce over the 16 lanes of each half-group (rows disjoint per half).
#pragma unroll
    for (int msk = 8; msk >= 1; msk >>= 1)
#pragma unroll
        for (int j = 0; j < 8; ++j)
            cmPart[j] += __shfl_xor(cmPart[j], msk, 32);

    if (lm == 0) {   // lanes 0 and 16: half-group leaders
        float cs = cscale[0];
#pragma unroll
        for (int j = 0; j < 8; ++j) {
            int m = half * 8 + j;
            int e = waveBase + m;
            if (e < E) {
                float f = cs * cmPart[j] / sDist[wid * 16 + m];
                int r = sR[wid * 16 + m];
                atomicAdd(&coordsOut[r * 3 + 0], f * sRel[wid * 16 + m][0]);
                atomicAdd(&coordsOut[r * 3 + 1], f * sRel[wid * 16 + m][1]);
                atomicAdd(&coordsOut[r * 3 + 2], f * sRel[wid * 16 + m][2]);
            }
        }
    }
}

// ---------------------------------------------------------------- node kernel
__global__ __launch_bounds__(256) void egnn_node(
    const float* __restrict__ h, const __bf16* __restrict__ hbf,
    const float* __restrict__ agg,
    const __bf16* __restrict__ w1s, const float* __restrict__ b1,
    const __bf16* __restrict__ w2s, const float* __restrict__ b2,
    const float* __restrict__ g, const float* __restrict__ beta,
    const float* __restrict__ fscale,
    float* __restrict__ hout, int N)
{
    __shared__ __bf16 sStage[2][16 * 512];
    __shared__ __bf16 sAct[8][16 * 256];

    const int tid  = threadIdx.x;
    const int wid  = tid >> 5;
    const int lane = tid & 31;
    const int lm   = lane & 15;
    const int half = lane >> 4;
    const int waveBase = blockIdx.x * 128 + wid * 16;

    int i = waveBase + lm;
    if (i >= N) i = N - 1;
    const __bf16* arow = hbf + (size_t)i * 256;
    const float*  grow = agg + (size_t)i * 256;

    v16bf aF[16];
#pragma unroll
    for (int c = 0; c < 8; ++c) {
        aF[c]     = ldA(arow, c * 32, half);
        aF[8 + c] = ldA_cvt(grow, c * 32, half);
    }

    // ---- GEMM1: [h | agg] @ n_w1 + b1, leaky -> sAct
    stage_issue(sStage[0], w1s, 0, 16, tid);
    for (int nt = 0; nt < 16; ++nt) {
        stage_next_and_wait<16>(sStage, w1s, nt, 16, tid);
        __syncthreads();
        const __bf16* buf = sStage[nt & 1];
        v8f acc = {0.f,0.f,0.f,0.f,0.f,0.f,0.f,0.f};
#pragma unroll
        for (int c = 0; c < 16; ++c)
            acc = wmma_bf16(aF[c], ldB_lds(buf, c, lm, half), acc);
        int n = nt * 16 + lm;
        float bb = b1[n];
#pragma unroll
        for (int j = 0; j < 8; ++j) {
            int m = half * 8 + j;
            sAct[wid][m * 256 + n] = (__bf16)leakyf(acc[j] + bb);
        }
        __syncthreads();
    }

    v16bf aF2[8];
    {
        const __bf16* ar = &sAct[wid][lm * 256];
#pragma unroll
        for (int c = 0; c < 8; ++c) aF2[c] = ldA(ar, c * 32, half);
    }

    // ---- GEMM2: y @ n_w2 + b2 -> sAct (no activation before LN)
    stage_issue(sStage[0], w2s, 0, 8, tid);
    for (int nt = 0; nt < 16; ++nt) {
        stage_next_and_wait<8>(sStage, w2s, nt, 16, tid);
        __syncthreads();
        const __bf16* buf = sStage[nt & 1];
        v8f acc = {0.f,0.f,0.f,0.f,0.f,0.f,0.f,0.f};
#pragma unroll
        for (int c = 0; c < 8; ++c)
            acc = wmma_bf16(aF2[c], ldB_lds(buf, c, lm, half), acc);
        int n = nt * 16 + lm;
        float bb = b2[n];
#pragma unroll
        for (int j = 0; j < 8; ++j) {
            int m = half * 8 + j;
            sAct[wid][m * 256 + n] = (__bf16)(acc[j] + bb);
        }
        __syncthreads();
    }

    // ---- LayerNorm + residual (wave-local)
    if (lane < 16 && (waveBase + lane) < N) {
        const __bf16* rowp = &sAct[wid][lane * 256];
        float mu = 0.f;
        for (int c2 = 0; c2 < 256; ++c2) mu += (float)rowp[c2];
        mu *= (1.0f / 256.0f);
        float var = 0.f;
        for (int c2 = 0; c2 < 256; ++c2) {
            float d = (float)rowp[c2] - mu; var += d * d;
        }
        var *= (1.0f / 256.0f);
        float rstd = rsqrtf(var + 1e-5f);
        float fs = fscale[0];
        size_t o = (size_t)(waveBase + lane) * 256;
        for (int c2 = 0; c2 < 256; ++c2) {
            float hu = ((float)rowp[c2] - mu) * rstd * g[c2] + beta[c2];
            hout[o + c2] = h[o + c2] + fs * hu;
        }
    }
}

// ---------------------------------------------------------------- launcher
extern "C" void kernel_launch(void* const* d_in, const int* in_sizes, int n_in,
                              void* d_out, int out_size, void* d_ws, size_t ws_size,
                              hipStream_t stream) {
    const float* h       = (const float*)d_in[0];
    const float* coords  = (const float*)d_in[1];
    const int*   ei      = (const int*)d_in[2];
    const float* e_w1    = (const float*)d_in[3];
    const float* e_b1    = (const float*)d_in[4];
    const float* e_w2    = (const float*)d_in[5];
    const float* e_b2    = (const float*)d_in[6];
    const float* e_g     = (const float*)d_in[7];
    const float* e_beta  = (const float*)d_in[8];
    const float* n_w1    = (const float*)d_in[9];
    const float* n_b1    = (const float*)d_in[10];
    const float* n_w2    = (const float*)d_in[11];
    const float* n_b2    = (const float*)d_in[12];
    const float* n_g     = (const float*)d_in[13];
    const float* n_beta  = (const float*)d_in[14];
    const float* c_w1    = (const float*)d_in[15];
    const float* c_b1    = (const float*)d_in[16];
    const float* c_w2    = (const float*)d_in[17];
    const float* cscale  = (const float*)d_in[18];
    const float* fscale  = (const float*)d_in[19];

    const int N = in_sizes[0] / 256;
    const int E = in_sizes[2] / 2;
    const int* rowI = ei;
    const int* colI = ei + E;

    char* ws = (char*)d_ws;
    size_t off = 0;
    auto alloc = [&](size_t bytes) -> void* {
        void* p = ws + off;
        off = (off + bytes + 255) & ~(size_t)255;
        return p;
    };
    __bf16* hbf  = (__bf16*)alloc((size_t)N * 256 * 2);
    __bf16* ew1s = (__bf16*)alloc((size_t)16 * 8192 * 2);
    __bf16* ew2s = (__bf16*)alloc((size_t)8 * 8192 * 2);
    __bf16* cw1s = (__bf16*)alloc((size_t)8 * 8192 * 2);
    __bf16* nw1s = (__bf16*)alloc((size_t)16 * 8192 * 2);
    __bf16* nw2s = (__bf16*)alloc((size_t)8 * 8192 * 2);
    float*  agg  = (float*)alloc((size_t)N * 256 * 4);
    (void)ws_size; (void)n_in; (void)out_size;

    float* hout = (float*)d_out;
    float* coordsOut = hout + (size_t)N * 256;

    const int T = 256;
    k_zero_f32<<<(N * 256 + T - 1) / T, T, 0, stream>>>(agg, N * 256);
    k_copy_f32<<<(N * 3 + T - 1) / T, T, 0, stream>>>(coordsOut, coords, N * 3);
    k_cvt_bf16<<<(N * 256 + T - 1) / T, T, 0, stream>>>(hbf, h, N * 256);
    k_swz<<<(16 * 8192 + T - 1) / T, T, 0, stream>>>(ew1s, e_w1, 16);
    k_swz<<<(8 * 8192 + T - 1) / T, T, 0, stream>>>(ew2s, e_w2, 8);
    k_swz<<<(8 * 8192 + T - 1) / T, T, 0, stream>>>(cw1s, c_w1, 8);
    k_swz<<<(16 * 8192 + T - 1) / T, T, 0, stream>>>(nw1s, n_w1, 16);
    k_swz<<<(8 * 8192 + T - 1) / T, T, 0, stream>>>(nw2s, n_w2, 8);

    egnn_edge<<<(E + 127) / 128, 256, 0, stream>>>(
        coords, rowI, colI, hbf, ew1s, e_w1, e_b1, ew2s, e_b2, e_g, e_beta,
        cw1s, c_b1, c_w2, cscale, agg, coordsOut, E);

    egnn_node<<<(N + 127) / 128, 256, 0, stream>>>(
        h, hbf, agg, nw1s, n_b1, nw2s, n_b2, n_g, n_beta, fscale, hout, N);
}
